// MoE_YTS_7138235646426
// MI455X (gfx1250) — matrix operand order repaired and checked by reading, not compiled, not run on hardware
//
#include <hip/hip_runtime.h>

typedef __attribute__((ext_vector_type(16))) _Float16 v16h;
typedef __attribute__((ext_vector_type(8)))  _Float16 v8h;
typedef __attribute__((ext_vector_type(8)))  float    v8f;

namespace {
constexpr int kB  = 32768;
constexpr int kD  = 256;
constexpr int kH1 = 512;
constexpr int kH2 = 256;
constexpr int kE  = 16;
constexpr int kO  = 64;

constexpr int TILE_B  = 64;     // rows per workgroup (4 M-tiles)
constexpr int THREADS = 256;    // 8 waves; each wave covers ALL M-tiles, an N-slice

// LDS row strides (halves), +8 pad keeps 16B alignment and staggers banks
constexpr int LDX  = kD  + 8;
constexpr int LDH1 = kH1 + 8;
constexpr int LDH2 = kH2 + 8;
}

// ---------------- prep: x fp32 -> f16 row-major ----------------
__global__ void cvt_x_f16(const float* __restrict__ x, _Float16* __restrict__ xh, int n) {
    int i = blockIdx.x * blockDim.x + threadIdx.x;
    if (i < n) xh[i] = (_Float16)x[i];
}

// ---------------- prep: pack W[e][K][N] fp32 into WMMA-B-native f16 layout --
// dst flat = (((e*NT + nt)*KS + ks)*32 + lane)*16 + i
// lane holds column n = nt*16 + lane%16, K = ks*32 + (lane>=16 ? 8 : 0) + (i&7) + (i>=8 ? 16 : 0)
__global__ void pack_b_f16(const float* __restrict__ w, _Float16* __restrict__ dst,
                           int K, int N, int total) {
    int f = blockIdx.x * blockDim.x + threadIdx.x;
    if (f >= total) return;
    int i    = f & 15;
    int lane = (f >> 4) & 31;
    int KS   = K >> 5;
    int NT   = N >> 4;
    int ks   = (f >> 9) % KS;
    int rest = (f >> 9) / KS;
    int nt   = rest % NT;
    int e    = rest / NT;
    int n = nt * 16 + (lane & 15);
    int k = ks * 32 + ((lane >> 4) << 3) + (i & 7) + ((i >> 3) << 4);
    dst[f] = (_Float16)w[((size_t)e * K + k) * N + n];
}

// ---------------- gating: softmax(x @ gate_w + gate_b) over E=16 ------------
__global__ void gates_kernel(const float* __restrict__ x, const float* __restrict__ gw,
                             const float* __restrict__ gb, float* __restrict__ gates) {
    __shared__ float gws[kD * kE];
    int tid = threadIdx.x;
    for (int i = tid; i < kD * kE; i += 256) gws[i] = gw[i];
    __syncthreads();
    int b = blockIdx.x * 256 + tid;
    float acc[kE];
#pragma unroll
    for (int e = 0; e < kE; ++e) acc[e] = gb[e];
    const float* xr = x + (size_t)b * kD;
    for (int k = 0; k < kD; ++k) {
        float xv = xr[k];
#pragma unroll
        for (int e = 0; e < kE; ++e) acc[e] += xv * gws[k * kE + e];
    }
    float m = acc[0];
#pragma unroll
    for (int e = 1; e < kE; ++e) m = fmaxf(m, acc[e]);
    float s = 0.f;
#pragma unroll
    for (int e = 0; e < kE; ++e) { acc[e] = __expf(acc[e] - m); s += acc[e]; }
    float inv = 1.f / s;
#pragma unroll
    for (int e = 0; e < kE; ++e) gates[(size_t)b * kE + e] = acc[e] * inv;
}

// ---------------- fragment loaders ----------------
// A fragment from a precomputed per-lane base (base already includes
// row = lane%16 and the half-wave K offset); ks offset folds into DS imm.
__device__ __forceinline__ v16h load_a_at(const _Float16* lane_base, int ks) {
    const _Float16* p = lane_base + ks * 32;
    v8h lo = *(const v8h*)p;
    v8h hi = *(const v8h*)(p + 16);
    return __builtin_shufflevector(lo, hi, 0,1,2,3,4,5,6,7,8,9,10,11,12,13,14,15);
}

// B fragment (32x16 f16) from prepacked global: lane-contiguous 32B chunk
__device__ __forceinline__ v16h load_b_frag(const _Float16* frag_base, int lane) {
    return *(const v16h*)(frag_base + lane * 16);
}

__device__ __forceinline__ v8f wmma_f16(v16h a, v16h b, v8f c) {
    return __builtin_amdgcn_wmma_f32_16x16x32_f16(false, a, false, b, (short)0, c, false, false);
}

// ---------------- fused MoE pipeline ----------------
// Each wave processes all 4 M-tiles (64 rows) for its own N-slice of every
// stage: every weight B-fragment is loaded once per block and feeds 4 WMMAs.
__global__ __launch_bounds__(THREADS, 1)
void moe_main(const _Float16* __restrict__ xh,
              const _Float16* __restrict__ w1p,
              const _Float16* __restrict__ w2p,
              const _Float16* __restrict__ w3p,
              const float* __restrict__ pb1,
              const float* __restrict__ pb2,
              const float* __restrict__ pb3,
              const float* __restrict__ gates,
              const float* __restrict__ head_w,
              const float* __restrict__ head_b,
              float* __restrict__ out) {
    __shared__ alignas(16) _Float16 sx [TILE_B * LDX];
    __shared__ alignas(16) _Float16 sh1[TILE_B * LDH1];
    __shared__ alignas(16) _Float16 sh2[TILE_B * LDH2];
    __shared__ float sg[TILE_B * kE];
    __shared__ float sred[8][TILE_B];       // per-wave partial head sums

    const int tid   = threadIdx.x;
    const int lane  = tid & 31;
    const int ng    = tid >> 5;             // wave index == N-slice id
    const int b0    = blockIdx.x * TILE_B;
    const int col   = lane & 15;            // N within 16-tile / A row
    const int half8 = (lane >> 4) << 3;     // C-layout M offset (0 or 8)

    // stage x tile (f16) and gates tile into LDS
    for (int c = tid; c < TILE_B * (kD / 8); c += THREADS) {
        int r  = c / (kD / 8);
        int cc = c % (kD / 8);
        v8h v = *(const v8h*)(xh + (size_t)(b0 + r) * kD + cc * 8);
        *(v8h*)(&sx[r * LDX + cc * 8]) = v;
    }
    for (int i = tid; i < TILE_B * kE; i += THREADS)
        sg[i] = gates[(size_t)b0 * kE + i];
    __syncthreads();

    // Per-M-tile A-fragment lane bases: DS addresses become base + small imm
    const _Float16* axb[4];
    const _Float16* a1b[4];
    const _Float16* a2b[4];
#pragma unroll
    for (int mi = 0; mi < 4; ++mi) {
        axb[mi] = sx  + (mi * 16 + col) * LDX  + half8;
        a1b[mi] = sh1 + (mi * 16 + col) * LDH1 + half8;
        a2b[mi] = sh2 + (mi * 16 + col) * LDH2 + half8;
    }
    const float* sgl = sg + half8 * kE;      // gate base for this lane half

    v8f accT[4];                             // [mi] persistent gated output acc
#pragma unroll
    for (int mi = 0; mi < 4; ++mi) accT[mi] = (v8f){};

#pragma unroll 1
    for (int e = 0; e < kE; ++e) {
        // ---- stage 1: sh1 = relu(x @ W1[e] + b1[e])  (K=256, N=512) ----
        // wave's N-tiles: ng*4 .. ng*4+3, processed as 2 pairs
        const _Float16* w1e = w1p + (size_t)e * (32 * 8 * 512);
#pragma unroll 1
        for (int np = 0; np < 2; ++np) {
            const int nt0 = ng * 4 + np * 2, nt1 = nt0 + 1;
            v8f c[4][2];
#pragma unroll
            for (int mi = 0; mi < 4; ++mi) { c[mi][0] = (v8f){}; c[mi][1] = (v8f){}; }
#pragma unroll 2
            for (int ks = 0; ks < 8; ++ks) {
                v16h bf0 = load_b_frag(w1e + ((size_t)nt0 * 8 + ks) * 512, lane);
                v16h bf1 = load_b_frag(w1e + ((size_t)nt1 * 8 + ks) * 512, lane);
                v16h a[4];
#pragma unroll
                for (int mi = 0; mi < 4; ++mi) a[mi] = load_a_at(axb[mi], ks);
#pragma unroll
                for (int mi = 0; mi < 4; ++mi) {
                    c[mi][0] = wmma_f16(a[mi], bf0, c[mi][0]);
                    c[mi][1] = wmma_f16(a[mi], bf1, c[mi][1]);
                }
            }
            const int n0 = nt0 * 16 + col, n1 = n0 + 16;
            const float bv0 = pb1[e * kH1 + n0];
            const float bv1 = pb1[e * kH1 + n1];
#pragma unroll
            for (int mi = 0; mi < 4; ++mi) {
                const int mb = mi * 16 + half8;
#pragma unroll
                for (int r = 0; r < 8; ++r) {
                    float h0 = c[mi][0][r] + bv0; h0 = h0 > 0.f ? h0 : 0.f;
                    float h1 = c[mi][1][r] + bv1; h1 = h1 > 0.f ? h1 : 0.f;
                    sh1[(mb + r) * LDH1 + n0] = (_Float16)h0;
                    sh1[(mb + r) * LDH1 + n1] = (_Float16)h1;
                }
            }
        }
        __syncthreads();

        // ---- stage 2: sh2 = relu(h1 @ W2[e] + b2[e])  (K=512, N=256) ----
        // wave's N-tiles: ng*2, ng*2+1 (one pair)
        const _Float16* w2e = w2p + (size_t)e * (16 * 16 * 512);
        {
            const int nt0 = ng * 2, nt1 = nt0 + 1;
            v8f c[4][2];
#pragma unroll
            for (int mi = 0; mi < 4; ++mi) { c[mi][0] = (v8f){}; c[mi][1] = (v8f){}; }
#pragma unroll 2
            for (int ks = 0; ks < 16; ++ks) {
                v16h bf0 = load_b_frag(w2e + ((size_t)nt0 * 16 + ks) * 512, lane);
                v16h bf1 = load_b_frag(w2e + ((size_t)nt1 * 16 + ks) * 512, lane);
                v16h a[4];
#pragma unroll
                for (int mi = 0; mi < 4; ++mi) a[mi] = load_a_at(a1b[mi], ks);
#pragma unroll
                for (int mi = 0; mi < 4; ++mi) {
                    c[mi][0] = wmma_f16(a[mi], bf0, c[mi][0]);
                    c[mi][1] = wmma_f16(a[mi], bf1, c[mi][1]);
                }
            }
            const int n0 = nt0 * 16 + col, n1 = n0 + 16;
            const float bv0 = pb2[e * kH2 + n0];
            const float bv1 = pb2[e * kH2 + n1];
#pragma unroll
            for (int mi = 0; mi < 4; ++mi) {
                const int mb = mi * 16 + half8;
#pragma unroll
                for (int r = 0; r < 8; ++r) {
                    float h0 = c[mi][0][r] + bv0; h0 = h0 > 0.f ? h0 : 0.f;
                    float h1 = c[mi][1][r] + bv1; h1 = h1 > 0.f ? h1 : 0.f;
                    sh2[(mb + r) * LDH2 + n0] = (_Float16)h0;
                    sh2[(mb + r) * LDH2 + n1] = (_Float16)h1;
                }
            }
        }
        __syncthreads();

        // ---- stage 3: eo = h2 @ W3[e]; accT += gate * (eo + b3[e]) ----
        // 4 N-tiles among 8 waves: nt = ng/2, K-half = ng&1.
        // Bias is added only by the even K-half so the gated sum stays exact.
        const _Float16* w3e = w3p + (size_t)e * (4 * 8 * 512);
        const int nt3 = ng >> 1;
        const int kb3 = (ng & 1) * 4;
        v8f eoc[4];
#pragma unroll
        for (int mi = 0; mi < 4; ++mi) eoc[mi] = (v8f){};
#pragma unroll 2
        for (int kk = 0; kk < 4; ++kk) {
            const int ks = kb3 + kk;
            v16h bf = load_b_frag(w3e + ((size_t)nt3 * 8 + ks) * 512, lane);
            v16h a[4];
#pragma unroll
            for (int mi = 0; mi < 4; ++mi) a[mi] = load_a_at(a2b[mi], ks);
#pragma unroll
            for (int mi = 0; mi < 4; ++mi)
                eoc[mi] = wmma_f16(a[mi], bf, eoc[mi]);
        }
        const float bv = (ng & 1) ? 0.f : pb3[e * kO + nt3 * 16 + col];
#pragma unroll
        for (int mi = 0; mi < 4; ++mi) {
#pragma unroll
            for (int r = 0; r < 8; ++r) {
                float gvr = sgl[(mi * 16 + r) * kE + e];
                accT[mi][r] += gvr * (eoc[mi][r] + bv);
            }
        }
    }

    // ---- head: yts[m] = sum_n accT[m][n]*head_w[n] + head_b ----
    // shfl-reduce within 16-lane halves, then combine the 8 wave slices in LDS
    const float hw = head_w[(ng >> 1) * 16 + col];
#pragma unroll
    for (int mi = 0; mi < 4; ++mi) {
#pragma unroll
        for (int r = 0; r < 8; ++r) {
            float v = accT[mi][r] * hw;
            v += __shfl_xor(v, 1, 32);   // xor<16 stays inside each 16-lane half
            v += __shfl_xor(v, 2, 32);
            v += __shfl_xor(v, 4, 32);
            v += __shfl_xor(v, 8, 32);
            if (col == 0) {
                int m = mi * 16 + half8 + r;
                sred[ng][m] = v;
            }
        }
    }
    __syncthreads();
    if (tid < TILE_B) {
        float v = sred[0][tid] + sred[1][tid] + sred[2][tid] + sred[3][tid]
                + sred[4][tid] + sred[5][tid] + sred[6][tid] + sred[7][tid];
        out[b0 + tid] = v + head_b[0];
    }
}

extern "C" void kernel_launch(void* const* d_in, const int* in_sizes, int n_in,
                              void* d_out, int out_size, void* d_ws, size_t ws_size,
                              hipStream_t stream) {
    const float* x      = (const float*)d_in[0];
    const float* gate_w = (const float*)d_in[1];
    const float* gate_b = (const float*)d_in[2];
    const float* W1     = (const float*)d_in[3];
    const float* b1     = (const float*)d_in[4];
    const float* W2     = (const float*)d_in[5];
    const float* b2     = (const float*)d_in[6];
    const float* W3     = (const float*)d_in[7];
    const float* b3     = (const float*)d_in[8];
    const float* head_w = (const float*)d_in[9];
    const float* head_b = (const float*)d_in[10];
    float* out = (float*)d_out;

    // workspace layout (≈26.5 MB): x-f16 | W1p | W2p | W3p | gates-f32
    _Float16* xh  = (_Float16*)d_ws;
    _Float16* w1p = xh  + (size_t)kB * kD;
    _Float16* w2p = w1p + (size_t)kE * kD  * kH1;
    _Float16* w3p = w2p + (size_t)kE * kH1 * kH2;
    float*    gts = (float*)(w3p + (size_t)kE * kH2 * kO);

    cvt_x_f16<<<(kB * kD) / 256, 256, 0, stream>>>(x, xh, kB * kD);
    pack_b_f16<<<(kE * kD  * kH1) / 256, 256, 0, stream>>>(W1, w1p, kD,  kH1, kE * kD  * kH1);
    pack_b_f16<<<(kE * kH1 * kH2) / 256, 256, 0, stream>>>(W2, w2p, kH1, kH2, kE * kH1 * kH2);
    pack_b_f16<<<(kE * kH2 * kO ) / 256, 256, 0, stream>>>(W3, w3p, kH2, kO,  kE * kH2 * kO);
    gates_kernel<<<kB / 256, 256, 0, stream>>>(x, gate_w, gate_b, gts);
    moe_main<<<kB / TILE_B, THREADS, 0, stream>>>(xh, w1p, w2p, w3p, b1, b2, b3,
                                                  gts, head_w, head_b, out);
}
